// DynamicsPredictor_54846732370389
// MI455X (gfx1250) — matrix operand order repaired and checked by reading, not compiled, not run on hardware
//
#include <hip/hip_runtime.h>
#include <hip/hip_bf16.h>

// ---------------------------------------------------------------------------
// Problem constants (from reference)
// ---------------------------------------------------------------------------
#define NP     2000
#define NN     2048
#define NREL   16384
#define NF     128
#define NINST  8
#define PIN    15      // particle input dim
#define PINP   32      // padded to K multiple of 32
#define RIN    44      // relation input dim
#define RINP   64      // padded

typedef __attribute__((ext_vector_type(16))) _Float16 v16h;
typedef __attribute__((ext_vector_type(8)))  float    v8f;

union F16x16 { v16h v; _Float16 h[16]; unsigned int u[8]; uint4 q[2]; };
union F32x8  { v8f  v; float f[8]; };

// ---------------------------------------------------------------------------
// Device-global scratch (self-contained; no dependence on ws_size)
// ---------------------------------------------------------------------------
__device__ int      g_recv[NREL];
__device__ int      g_send[NREL];
__device__ float    g_snorm[NN * 9];
__device__ float    g_pinf [NN * PIN];
__device__ _Float16 g_pin16[NN * PINP];
__device__ _Float16 g_rel16[NREL * RINP];

// Weights stored pre-swizzled into B-fragment order: [K/32][8 ntile][32 lane][16 h]
__device__ _Float16 g_wpe0[PINP * NF];
__device__ _Float16 g_wpe1[NF * NF];
__device__ _Float16 g_wpe2[NF * NF];
__device__ _Float16 g_wre0[RINP * NF];
__device__ _Float16 g_wre1[NF * NF];
__device__ _Float16 g_wre2[NF * NF];
__device__ _Float16 g_wpp [2 * NF * NF];
__device__ _Float16 g_wrp [3 * NF * NF];
__device__ _Float16 g_wnp0[NF * NF];
__device__ _Float16 g_wnp1[NF * NF];
__device__ _Float16 g_wnp2[NF * NF];   // np_w2 padded 128x3 -> 128x128
__device__ float    g_bnp2[NF];        // np_b2 padded 3 -> 128

__device__ _Float16 g_h16a[NREL * NF];  // f16 hidden ping
__device__ _Float16 g_h16b[NREL * NF];  // f16 hidden pong
__device__ float    g_penc[NN * NF];    // particle_encode (f32)
__device__ float    g_renc[NREL * NF];  // relation_encode (f32)
__device__ float    g_effa[NN * NF];
__device__ float    g_effb[NN * NF];
__device__ _Float16 g_relcat[NREL * 3 * NF];
__device__ float    g_effrel[NREL * NF];
__device__ float    g_agg [NN * NF];
__device__ _Float16 g_pcat[NN * 2 * NF];
__device__ _Float16 g_np16[NN * NF];
__device__ float    g_npout[NN * NF];

// ---------------------------------------------------------------------------
// One-hot -> index decode (bandwidth-critical: 268MB total, use b128 loads)
// ---------------------------------------------------------------------------
__global__ void extract_idx_k(const float4* __restrict__ onehot, int* __restrict__ idx)
{
    int row = blockIdx.x;                      // 16384 rows
    const float4* p = onehot + (size_t)row * (NN / 4);
    for (int c = threadIdx.x; c < NN / 4; c += blockDim.x) {
        float4 v = p[c];
        if (v.x > 0.5f) idx[row] = 4 * c + 0;
        if (v.y > 0.5f) idx[row] = 4 * c + 1;
        if (v.z > 0.5f) idx[row] = 4 * c + 2;
        if (v.w > 0.5f) idx[row] = 4 * c + 3;
    }
}

// ---------------------------------------------------------------------------
// Per-particle feature build: state residuals, p_inputs (f32 + padded f16)
// ---------------------------------------------------------------------------
__global__ void build_particle_k(const float* __restrict__ state,  // [3,NN,3]
                                 const float* __restrict__ attrs,  // [NN,2]
                                 const float* __restrict__ action, // [NN,3]
                                 const float* __restrict__ phys,   // [NP]
                                 float* __restrict__ pinf,
                                 _Float16* __restrict__ pin16,
                                 float* __restrict__ snorm)
{
    int n = blockIdx.x * 256 + threadIdx.x;
    if (n >= NN) return;
    float s[3][3];
    #pragma unroll
    for (int t = 0; t < 3; ++t)
        #pragma unroll
        for (int d = 0; d < 3; ++d)
            s[t][d] = state[((size_t)t * NN + n) * 3 + d];
    float sn[9];
    #pragma unroll
    for (int d = 0; d < 3; ++d) {
        sn[d]     = s[1][d] - s[0][d];
        sn[3 + d] = s[2][d] - s[1][d];
        sn[6 + d] = s[2][d];
    }
    #pragma unroll
    for (int k = 0; k < 9; ++k) snorm[(size_t)n * 9 + k] = sn[k];

    float pv[PIN];
    pv[0] = attrs[n * 2 + 0];
    pv[1] = attrs[n * 2 + 1];
    #pragma unroll
    for (int k = 0; k < 9; ++k) pv[2 + k] = sn[k];
    pv[11] = (n < NP) ? phys[n] : 0.0f;
    #pragma unroll
    for (int d = 0; d < 3; ++d) pv[12 + d] = action[n * 3 + d];

    #pragma unroll
    for (int c = 0; c < PIN; ++c) pinf[(size_t)n * PIN + c] = pv[c];
    #pragma unroll
    for (int c = 0; c < PINP; ++c)
        pin16[(size_t)n * PINP + c] = (c < PIN) ? (_Float16)pv[c] : (_Float16)0.0f;
}

// ---------------------------------------------------------------------------
// Per-relation feature build (replaces Rr@X / Rs@X dense products by gathers)
// ---------------------------------------------------------------------------
__global__ void build_rel_k(const int* __restrict__ recv, const int* __restrict__ send,
                            const float* __restrict__ pinf, const float* __restrict__ attrs,
                            const float* __restrict__ pinst, const float* __restrict__ snorm,
                            _Float16* __restrict__ rel16)
{
    int r = blockIdx.x * 256 + threadIdx.x;
    if (r >= NREL) return;
    int i = recv[r], j = send[r];
    _Float16* o = rel16 + (size_t)r * RINP;
    int idx = 0;
    #pragma unroll
    for (int c = 0; c < PIN; ++c) o[idx++] = (_Float16)pinf[(size_t)i * PIN + c];
    #pragma unroll
    for (int c = 0; c < PIN; ++c) o[idx++] = (_Float16)pinf[(size_t)j * PIN + c];
    o[idx++] = (_Float16)attrs[i * 2 + 0];
    o[idx++] = (_Float16)attrs[i * 2 + 1];
    o[idx++] = (_Float16)attrs[j * 2 + 0];
    o[idx++] = (_Float16)attrs[j * 2 + 1];
    float gd = 0.0f;
    #pragma unroll
    for (int k = 0; k < NINST; ++k) {
        float gi = (i < NP) ? pinst[(size_t)i * NINST + k] : 0.0f;
        float gj = (j < NP) ? pinst[(size_t)j * NINST + k] : 0.0f;
        gd += fabsf(gi - gj);
    }
    o[idx++] = (_Float16)gd;
    #pragma unroll
    for (int c = 0; c < 9; ++c)
        o[idx++] = (_Float16)(snorm[(size_t)i * 9 + c] - snorm[(size_t)j * 9 + c]);
    while (idx < RINP) o[idx++] = (_Float16)0.0f;
}

// ---------------------------------------------------------------------------
// Weight f32 [K,N] -> f16 swizzled into WMMA B-fragment order:
//   dst[kc][nt][lane][e]  (kc = K/32 chunks, nt = 8 col-tiles, 32 lanes, 16 h)
//   lane = lhi*16+lrow -> col = nt*16+lrow,  k = kc*32 + lhi*16 + e
// So a lane's whole 32B fragment half is contiguous -> 2x global_load_b128.
// ---------------------------------------------------------------------------
__global__ void swizzle_weight_k(const float* __restrict__ src, _Float16* __restrict__ dst,
                                 int K, int N, int Kp)
{
    int i = blockIdx.x * 256 + threadIdx.x;
    if (i >= Kp * NF) return;
    int e    = i & 15;
    int lane = (i >> 4) & 31;
    int nt   = (i >> 9) & 7;
    int kc   = i >> 12;
    int k    = kc * 32 + (lane >> 4) * 16 + e;
    int col  = nt * 16 + (lane & 15);
    float v = (k < K && col < N) ? src[(size_t)k * N + col] : 0.0f;
    dst[i] = (_Float16)v;
}

__global__ void pad_bias_k(const float* __restrict__ src, float* __restrict__ dst, int N)
{
    int c = threadIdx.x;
    if (c < NF) dst[c] = (c < N) ? src[c] : 0.0f;
}

// ---------------------------------------------------------------------------
// WMMA GEMM: C[M,128] = act( A[M,K](f16) @ W(swizzled) + bias + resid )
// Block: 256 threads = 8 waves, 128x128 tile. Wave: 32x64 (2x4 WMMA frags).
// A staged to LDS via GLOBAL_LOAD_ASYNC_TO_LDS_B128 (ASYNCcnt); A frags are
// 2x ds_load_b128; B frags are 2x coalesced global_load_b128 from swizzled W.
// M multiple of 128; K multiple of 32.
// ---------------------------------------------------------------------------
#define SP 40   // LDS pitch in halves: 80B rows -> 16B aligned, conflict-free

__launch_bounds__(256)
__global__ void gemm_k(const _Float16* __restrict__ A, int K,
                       const _Float16* __restrict__ W,
                       const float* __restrict__ bias,
                       const float* __restrict__ resid,
                       float* __restrict__ Cf32,
                       _Float16* __restrict__ Cf16,
                       int relu)
{
    __shared__ __align__(16) _Float16 sA[128 * SP];  // [row][k] 32 halves used

    const int tid  = threadIdx.x;
    const int lane = tid & 31;
    const int wv   = tid >> 5;
    const int lrow = lane & 15;
    const int lhi  = lane >> 4;
    const size_t m0 = (size_t)blockIdx.x * 128;

    const int wm = (wv & 3) * 32;   // wave M offset in tile
    const int wn = (wv >> 2) * 64;  // wave N offset in tile

    F32x8 acc[2][4];
    #pragma unroll
    for (int a = 0; a < 2; ++a)
        #pragma unroll
        for (int b = 0; b < 4; ++b)
            #pragma unroll
            for (int r = 0; r < 8; ++r) acc[a][b].f[r] = 0.0f;

    for (int k0 = 0; k0 < K; k0 += 32) {
        // B fragments: direct from pre-swizzled W (L2-resident, coalesced)
        const int kc = k0 >> 5;
        F16x16 bf[4];
        #pragma unroll
        for (int nt = 0; nt < 4; ++nt) {
            const _Float16* wf =
                W + ((size_t)((kc * 8) + (wn >> 4) + nt) * 32 + lane) * 16;
            bf[nt].q[0] = *(const uint4*)(wf);
            bf[nt].q[1] = *(const uint4*)(wf + 8);
        }

        __syncthreads();   // previous chunk's LDS reads complete

        // Async-stage A chunk: 128 rows x 32 halves = 512 x 16B segments
        #pragma unroll
        for (int j = 0; j < 2; ++j) {
            int idx = tid + 256 * j;
            int r = idx >> 2, q = idx & 3;
            const _Float16* gp = &A[(m0 + r) * (size_t)K + k0 + q * 8];
            unsigned lp = (unsigned)(unsigned long long)(uintptr_t)&sA[r * SP + q * 8];
            asm volatile("global_load_async_to_lds_b128 %0, %1, off"
                         :: "v"(lp), "v"((unsigned long long)(uintptr_t)gp)
                         : "memory");
        }
        asm volatile("s_wait_asynccnt 0x0" ::: "memory");
        __syncthreads();   // all waves' async writes visible

        // A fragments (contiguous k order, matching swizzled B): K run
        // [kb,kb+16) with kb = 16*lhi is contiguous -> 2x ds_load_b128
        F16x16 af[2];
        #pragma unroll
        for (int mt = 0; mt < 2; ++mt) {
            const int row = wm + mt * 16 + lrow;
            const int kb  = lhi * 16;
            af[mt].q[0] = *(const uint4*)&sA[row * SP + kb];
            af[mt].q[1] = *(const uint4*)&sA[row * SP + kb + 8];
        }

        #pragma unroll
        for (int mt = 0; mt < 2; ++mt)
            #pragma unroll
            for (int nt = 0; nt < 4; ++nt)
                acc[mt][nt].v = __builtin_amdgcn_wmma_f32_16x16x32_f16(
                    false, af[mt].v, false, bf[nt].v,
                    (short)0, acc[mt][nt].v, false, false);
    }

    // Epilogue: D layout -> lane col = lrow, VGPR r -> row = r + 8*lhi
    #pragma unroll
    for (int mt = 0; mt < 2; ++mt)
        #pragma unroll
        for (int nt = 0; nt < 4; ++nt)
            #pragma unroll
            for (int r = 0; r < 8; ++r) {
                size_t row = m0 + wm + mt * 16 + r + 8 * lhi;
                int    col = wn + nt * 16 + lrow;
                float  v   = acc[mt][nt].f[r];
                if (bias)  v += bias[col];
                if (resid) v += resid[row * NF + col];
                if (relu)  v = fmaxf(v, 0.0f);
                if (Cf32)  Cf32[row * NF + col] = v;
                if (Cf16)  Cf16[row * NF + col] = (_Float16)v;
            }
}

// ---------------------------------------------------------------------------
// Propagation-step helpers
// ---------------------------------------------------------------------------
__global__ void rel_concat_k(const float* __restrict__ relenc, const float* __restrict__ eff,
                             const int* __restrict__ recv, const int* __restrict__ send,
                             _Float16* __restrict__ out)  // [NREL, 384]
{
    int r = blockIdx.x;
    int c = threadIdx.x;   // 128
    int i = recv[r], j = send[r];
    _Float16* o = out + (size_t)r * (3 * NF);
    o[c]            = (_Float16)relenc[(size_t)r * NF + c];
    o[NF + c]       = (_Float16)eff[(size_t)i * NF + c];
    o[2 * NF + c]   = (_Float16)eff[(size_t)j * NF + c];
}

__global__ void zero_f32_k(float* __restrict__ p, int n)
{
    int i = blockIdx.x * 256 + threadIdx.x;
    if (i < n) p[i] = 0.0f;
}

__global__ void scatter_agg_k(const float* __restrict__ effrel,
                              const int* __restrict__ recv, float* __restrict__ agg)
{
    int t = blockIdx.x * 256 + threadIdx.x;   // NREL*NF
    int r = t >> 7, c = t & (NF - 1);
    atomicAdd(&agg[(size_t)recv[r] * NF + c], effrel[t]);
}

__global__ void part_concat_k(const float* __restrict__ penc, const float* __restrict__ agg,
                              _Float16* __restrict__ out)   // [NN, 256]
{
    int t = blockIdx.x * 256 + threadIdx.x;   // NN*NF
    int n = t >> 7, c = t & (NF - 1);
    out[(size_t)n * (2 * NF) + c]      = (_Float16)penc[t];
    out[(size_t)n * (2 * NF) + NF + c] = (_Float16)agg[t];
}

__global__ void cvt_f16_k(const float* __restrict__ src, _Float16* __restrict__ dst, int n)
{
    int i = blockIdx.x * 256 + threadIdx.x;
    if (i < n) dst[i] = (_Float16)src[i];
}

__global__ void finalize_k(const float* __restrict__ npout, const float* __restrict__ state,
                           float* __restrict__ out)
{
    int t = blockIdx.x * 256 + threadIdx.x;   // NP*3
    if (t >= NP * 3) return;
    int n = t / 3, d = t - n * 3;
    float m  = npout[(size_t)n * NF + d];
    float mc = fminf(fmaxf(m, -100.0f), 100.0f);
    out[t]          = state[((size_t)2 * NN + n) * 3 + d] + mc;  // pred_pos
    out[NP * 3 + t] = m;                                         // pred_motion
}

// ---------------------------------------------------------------------------
// Launcher
// ---------------------------------------------------------------------------
static void* sym_addr(const void* s)
{
    void* p = nullptr;
    (void)hipGetSymbolAddress(&p, s);
    return p;
}

extern "C" void kernel_launch(void* const* d_in, const int* in_sizes, int n_in,
                              void* d_out, int out_size, void* d_ws, size_t ws_size,
                              hipStream_t stream)
{
    (void)in_sizes; (void)n_in; (void)d_ws; (void)ws_size; (void)out_size;

    const float* state  = (const float*)d_in[0];
    const float* attrs  = (const float*)d_in[1];
    const float* Rr     = (const float*)d_in[2];
    const float* Rs     = (const float*)d_in[3];
    const float* pinst  = (const float*)d_in[4];
    const float* action = (const float*)d_in[5];
    const float* phys   = (const float*)d_in[6];
    const float* pe_w0 = (const float*)d_in[7],  * pe_b0 = (const float*)d_in[8];
    const float* pe_w1 = (const float*)d_in[9],  * pe_b1 = (const float*)d_in[10];
    const float* pe_w2 = (const float*)d_in[11], * pe_b2 = (const float*)d_in[12];
    const float* re_w0 = (const float*)d_in[13], * re_b0 = (const float*)d_in[14];
    const float* re_w1 = (const float*)d_in[15], * re_b1 = (const float*)d_in[16];
    const float* re_w2 = (const float*)d_in[17], * re_b2 = (const float*)d_in[18];
    const float* pp_w  = (const float*)d_in[19], * pp_b  = (const float*)d_in[20];
    const float* rp_w  = (const float*)d_in[21], * rp_b  = (const float*)d_in[22];
    const float* np_w0 = (const float*)d_in[23], * np_b0 = (const float*)d_in[24];
    const float* np_w1 = (const float*)d_in[25], * np_b1 = (const float*)d_in[26];
    const float* np_w2 = (const float*)d_in[27], * np_b2 = (const float*)d_in[28];

    int*      recv   = (int*)     sym_addr(g_recv);
    int*      send   = (int*)     sym_addr(g_send);
    float*    snorm  = (float*)   sym_addr(g_snorm);
    float*    pinf   = (float*)   sym_addr(g_pinf);
    _Float16* pin16  = (_Float16*)sym_addr(g_pin16);
    _Float16* rel16  = (_Float16*)sym_addr(g_rel16);
    _Float16* wpe0   = (_Float16*)sym_addr(g_wpe0);
    _Float16* wpe1   = (_Float16*)sym_addr(g_wpe1);
    _Float16* wpe2   = (_Float16*)sym_addr(g_wpe2);
    _Float16* wre0   = (_Float16*)sym_addr(g_wre0);
    _Float16* wre1   = (_Float16*)sym_addr(g_wre1);
    _Float16* wre2   = (_Float16*)sym_addr(g_wre2);
    _Float16* wpp    = (_Float16*)sym_addr(g_wpp);
    _Float16* wrp    = (_Float16*)sym_addr(g_wrp);
    _Float16* wnp0   = (_Float16*)sym_addr(g_wnp0);
    _Float16* wnp1   = (_Float16*)sym_addr(g_wnp1);
    _Float16* wnp2   = (_Float16*)sym_addr(g_wnp2);
    float*    bnp2   = (float*)   sym_addr(g_bnp2);
    _Float16* h16a   = (_Float16*)sym_addr(g_h16a);
    _Float16* h16b   = (_Float16*)sym_addr(g_h16b);
    float*    penc   = (float*)   sym_addr(g_penc);
    float*    renc   = (float*)   sym_addr(g_renc);
    float*    effa   = (float*)   sym_addr(g_effa);
    float*    effb   = (float*)   sym_addr(g_effb);
    _Float16* relcat = (_Float16*)sym_addr(g_relcat);
    float*    effrel = (float*)   sym_addr(g_effrel);
    float*    agg    = (float*)   sym_addr(g_agg);
    _Float16* pcat   = (_Float16*)sym_addr(g_pcat);
    _Float16* np16   = (_Float16*)sym_addr(g_np16);
    float*    npout  = (float*)   sym_addr(g_npout);

    // 1. decode one-hot Rr/Rs -> gather indices (bandwidth-dominant step)
    extract_idx_k<<<NREL, 256, 0, stream>>>((const float4*)Rr, recv);
    extract_idx_k<<<NREL, 256, 0, stream>>>((const float4*)Rs, send);

    // 2. feature builds
    build_particle_k<<<NN / 256, 256, 0, stream>>>(state, attrs, action, phys,
                                                   pinf, pin16, snorm);
    build_rel_k<<<NREL / 256, 256, 0, stream>>>(recv, send, pinf, attrs, pinst,
                                                snorm, rel16);

    // 3. weight conversion to f16, swizzled into B-fragment order
    swizzle_weight_k<<<(PINP * NF) / 256, 256, 0, stream>>>(pe_w0, wpe0, PIN, NF, PINP);
    swizzle_weight_k<<<(NF * NF) / 256,  256, 0, stream>>>(pe_w1, wpe1, NF, NF, NF);
    swizzle_weight_k<<<(NF * NF) / 256,  256, 0, stream>>>(pe_w2, wpe2, NF, NF, NF);
    swizzle_weight_k<<<(RINP * NF) / 256, 256, 0, stream>>>(re_w0, wre0, RIN, NF, RINP);
    swizzle_weight_k<<<(NF * NF) / 256,  256, 0, stream>>>(re_w1, wre1, NF, NF, NF);
    swizzle_weight_k<<<(NF * NF) / 256,  256, 0, stream>>>(re_w2, wre2, NF, NF, NF);
    swizzle_weight_k<<<(2 * NF * NF) / 256, 256, 0, stream>>>(pp_w, wpp, 2 * NF, NF, 2 * NF);
    swizzle_weight_k<<<(3 * NF * NF) / 256, 256, 0, stream>>>(rp_w, wrp, 3 * NF, NF, 3 * NF);
    swizzle_weight_k<<<(NF * NF) / 256,  256, 0, stream>>>(np_w0, wnp0, NF, NF, NF);
    swizzle_weight_k<<<(NF * NF) / 256,  256, 0, stream>>>(np_w1, wnp1, NF, NF, NF);
    swizzle_weight_k<<<(NF * NF) / 256,  256, 0, stream>>>(np_w2, wnp2, NF, 3, NF);
    pad_bias_k<<<1, 128, 0, stream>>>(np_b2, bnp2, 3);

    // 4. particle encoder (M=2048): 15->128->128->128, relu each layer
    gemm_k<<<NN / 128, 256, 0, stream>>>(pin16, PINP, wpe0, pe_b0, nullptr, nullptr, h16a, 1);
    gemm_k<<<NN / 128, 256, 0, stream>>>(h16a,  NF,   wpe1, pe_b1, nullptr, nullptr, h16b, 1);
    gemm_k<<<NN / 128, 256, 0, stream>>>(h16b,  NF,   wpe2, pe_b2, nullptr, penc, nullptr, 1);

    // 5. relation encoder (M=16384): 44->128->128->128
    gemm_k<<<NREL / 128, 256, 0, stream>>>(rel16, RINP, wre0, re_b0, nullptr, nullptr, h16a, 1);
    gemm_k<<<NREL / 128, 256, 0, stream>>>(h16a,  NF,   wre1, re_b1, nullptr, nullptr, h16b, 1);
    gemm_k<<<NREL / 128, 256, 0, stream>>>(h16b,  NF,   wre2, re_b2, nullptr, renc, nullptr, 1);

    // 6. propagation (3 steps). effect ping-pong: penc -> effa -> effb -> effa
    const float* effsrc = penc;
    float*       effdst = effa;
    for (int it = 0; it < 3; ++it) {
        rel_concat_k<<<NREL, 128, 0, stream>>>(renc, effsrc, recv, send, relcat);
        gemm_k<<<NREL / 128, 256, 0, stream>>>(relcat, 3 * NF, wrp, rp_b,
                                               nullptr, effrel, nullptr, 1);
        zero_f32_k<<<(NN * NF) / 256, 256, 0, stream>>>(agg, NN * NF);
        scatter_agg_k<<<(NREL * NF) / 256, 256, 0, stream>>>(effrel, recv, agg);
        part_concat_k<<<(NN * NF) / 256, 256, 0, stream>>>(penc, agg, pcat);
        gemm_k<<<NN / 128, 256, 0, stream>>>(pcat, 2 * NF, wpp, pp_b,
                                             effsrc, effdst, nullptr, 1);
        effsrc = effdst;
        effdst = (it == 0) ? effb : effa;
    }
    // final effect is in effa

    // 7. predictor head: 128->128->128->3 (padded to 128), no relu on last
    cvt_f16_k<<<(NN * NF) / 256, 256, 0, stream>>>(effa, np16, NN * NF);
    gemm_k<<<NN / 128, 256, 0, stream>>>(np16, NF, wnp0, np_b0, nullptr, nullptr, h16a, 1);
    gemm_k<<<NN / 128, 256, 0, stream>>>(h16a, NF, wnp1, np_b1, nullptr, nullptr, h16b, 1);
    gemm_k<<<NN / 128, 256, 0, stream>>>(h16b, NF, wnp2, bnp2, nullptr, npout, nullptr, 0);

    // 8. clamp + add last position, write (pred_pos, pred_motion)
    finalize_k<<<(NP * 3 + 255) / 256, 256, 0, stream>>>(npout, state, (float*)d_out);
}
